// AttentionBlock_60662118088881
// MI455X (gfx1250) — matrix operand order repaired and checked by reading, not compiled, not run on hardware
//
#include <hip/hip_runtime.h>
#include <stdint.h>

#define B_   4
#define T_   4096
#define DIN_ 1024
#define DK_  64

typedef __attribute__((ext_vector_type(16))) __bf16 bf16x16;
typedef __attribute__((ext_vector_type(8)))  float  v8f;
typedef int v4i_t __attribute__((vector_size(16)));   // matches async-LDS builtin param

union U16 {
    bf16x16 v;
    uint4 qq[2];
    unsigned short u[16];
    __bf16 h[16];
};

// Native f32->bf16 (RNE) — emits v_cvt_pk_bf16_f32 class ops.
__device__ __forceinline__ unsigned short f2bf(float f) {
    __bf16 h = (__bf16)f;
    return __builtin_bit_cast(unsigned short, h);
}

#if defined(__AMDGCN__) && __has_builtin(__builtin_amdgcn_global_load_async_to_lds_b128) && \
    __has_builtin(__builtin_amdgcn_s_wait_asynccnt)
#define ASYNC_LDS 1
#define GAS(p) ((__attribute__((address_space(1))) v4i_t*)(p))
#define LAS(p) ((__attribute__((address_space(3))) v4i_t*)(p))
#else
#define ASYNC_LDS 0
#endif

// Cross-lane xor-butterfly within 16-lane rows.  v_permlane16_b32 is pure
// VALU (no LDS round-trip / dscnt wait, unlike ds_bpermute that __shfl_xor
// lowers to).  Selector nibble for lane j is (j ^ MASK).
#if defined(__AMDGCN__) && __has_builtin(__builtin_amdgcn_permlane16)
#define HAVE_PERMLANE 1
#else
#define HAVE_PERMLANE 0
#endif

template <int MASK>
__device__ __forceinline__ float lane_xor16(float v) {
#if HAVE_PERMLANE
    constexpr unsigned rep = (unsigned)MASK * 0x11111111u;
    constexpr unsigned lo  = 0x76543210u ^ rep;
    constexpr unsigned hi  = 0xFEDCBA98u ^ rep;
    unsigned i = __builtin_bit_cast(unsigned, v);
    i = __builtin_amdgcn_permlane16(i, i, lo, hi, false, false);
    return __builtin_bit_cast(float, i);
#else
    return __shfl_xor(v, MASK);
#endif
}

__device__ __forceinline__ float rowmax16(float v) {
    v = fmaxf(v, lane_xor16<1>(v));
    v = fmaxf(v, lane_xor16<2>(v));
    v = fmaxf(v, lane_xor16<4>(v));
    v = fmaxf(v, lane_xor16<8>(v));
    return v;
}

__device__ __forceinline__ float rowsum16(float v) {
    v += lane_xor16<1>(v);
    v += lane_xor16<2>(v);
    v += lane_xor16<4>(v);
    v += lane_xor16<8>(v);
    return v;
}

// ---------------------------------------------------------------------------
// Kernel 0: W [1024][64] f32  ->  Wt [64][1024] bf16 (k-contiguous)
// ---------------------------------------------------------------------------
__global__ void prep_wt(const float* __restrict__ W, unsigned short* __restrict__ Wt) {
    int idx = blockIdx.x * 256 + threadIdx.x;   // over 64*1024
    if (idx >= 64 * 1024) return;
    int n = idx >> 10;
    int k = idx & 1023;
    Wt[idx] = f2bf(W[k * 64 + n]);
}

// ---------------------------------------------------------------------------
// Kernel 1: one projection GEMM [16384x1024]*[1024x64] in bf16 WMMA.
// One wave per 16-row tile; N=64 = 4 tiles of 16; K-loop 32 chunks of 32.
// MODE 0/1: write [row][64]; MODE 2: write transposed [B][64][T].
// ---------------------------------------------------------------------------
template <int MODE>
__global__ __launch_bounds__(32) void qkv_gemm(
        const float* __restrict__ x, const unsigned short* __restrict__ W,
        unsigned short* __restrict__ dst) {
    const int m0   = blockIdx.x * 16;
    const int lane = threadIdx.x;
    const int half = lane >> 4;
    const int lp   = lane & 15;

    v8f acc[4] = {};
    const float* xrow = x + (size_t)(m0 + lp) * DIN_;   // A row = lane%16

    for (int kc = 0; kc < 32; ++kc) {
        const int kb = kc * 32;
        // A operand: K = 16*half + [0..15]  (contiguous)
        U16 a;
        float4 f0 = *(const float4*)(xrow + kb + 16 * half);
        float4 f1 = *(const float4*)(xrow + kb + 16 * half + 4);
        float4 f2 = *(const float4*)(xrow + kb + 16 * half + 8);
        float4 f3 = *(const float4*)(xrow + kb + 16 * half + 12);
        a.h[0]  = (__bf16)f0.x; a.h[1]  = (__bf16)f0.y; a.h[2]  = (__bf16)f0.z; a.h[3]  = (__bf16)f0.w;
        a.h[4]  = (__bf16)f1.x; a.h[5]  = (__bf16)f1.y; a.h[6]  = (__bf16)f1.z; a.h[7]  = (__bf16)f1.w;
        a.h[8]  = (__bf16)f2.x; a.h[9]  = (__bf16)f2.y; a.h[10] = (__bf16)f2.z; a.h[11] = (__bf16)f2.w;
        a.h[12] = (__bf16)f3.x; a.h[13] = (__bf16)f3.y; a.h[14] = (__bf16)f3.z; a.h[15] = (__bf16)f3.w;
        #pragma unroll
        for (int nt = 0; nt < 4; ++nt) {
            // B operand: N = nt*16+lp, K = kb + 16*half + e  (contiguous 16 bf16)
            bf16x16 bv = *(const bf16x16*)(W + (nt * 16 + lp) * 1024 + kb + 16 * half);
            acc[nt] = __builtin_amdgcn_wmma_f32_16x16x32_bf16(
                false, a.v, false, bv, (short)0, acc[nt], false, false);
        }
    }

    #pragma unroll
    for (int nt = 0; nt < 4; ++nt) {
        #pragma unroll
        for (int r = 0; r < 8; ++r) {
            const int row = m0 + r + 8 * half;    // C: M = r + 8*(lane/16)
            const int col = nt * 16 + lp;         // C: N = lane%16
            unsigned short hv = f2bf(acc[nt][r]);
            if constexpr (MODE < 2) {
                dst[(size_t)row * 64 + col] = hv;
            } else {
                const int bb = row >> 12;         // T_ = 4096
                const int t  = row & 4095;
                dst[((size_t)bb * 64 + col) * T_ + t] = hv;
            }
        }
    }
}

// ---------------------------------------------------------------------------
// Kernel 2: fused flash-style attention with the reference's quirks:
//   softmax over ALL keys, masked positions forced to bf16 -inf in P (the
//   P*V WMMA then reproduces inf/NaN propagation), final *(1/8) scale.
// 4 waves/WG, each owns 16 query rows; 64-key K/V tiles staged in LDS
// via gfx1250 async-to-LDS loads.
// ---------------------------------------------------------------------------
__global__ __launch_bounds__(128) void attn_kernel(
        const unsigned short* __restrict__ Qb, const unsigned short* __restrict__ Kb,
        const unsigned short* __restrict__ Vt, float* __restrict__ out) {
    __shared__ __align__(32) unsigned short Kt_s[64 * 64];   // [key][dk]
    __shared__ __align__(32) unsigned short Vt_s[64 * 64];   // [dv][key]
    __shared__ __align__(32) unsigned short P_s[4 * 16 * 32];

    const int b     = blockIdx.y;
    const int q0    = blockIdx.x * 64;
    const int tid   = threadIdx.x;
    const int wave  = tid >> 5;
    const int lane  = tid & 31;
    const int half  = lane >> 4;
    const int lp    = lane & 15;
    const int qbase = q0 + wave * 16;

    // Q tile into registers as two A operands (dk chunks 0-31, 32-63)
    U16 qa[2];
    #pragma unroll
    for (int c = 0; c < 2; ++c) {
        const unsigned short* p = Qb + ((size_t)b * T_ + qbase + lp) * 64 + c * 32;
        qa[c].qq[0] = *(const uint4*)(p + 16 * half);
        qa[c].qq[1] = *(const uint4*)(p + 16 * half + 8);
    }

    v8f o[4] = {};
    float m_r[8], l_r[8];
    #pragma unroll
    for (int r = 0; r < 8; ++r) { m_r[r] = -INFINITY; l_r[r] = 0.0f; }

    unsigned short* Pw = P_s + wave * 512;

    for (int kt = 0; kt < T_; kt += 64) {
        __syncthreads();
        {   // stage K tile (8KB contiguous) and V tile (64 rows of 128B) into LDS
            const unsigned short* gK = Kb + ((size_t)b * T_ + kt) * 64;
#if ASYNC_LDS
            #pragma unroll
            for (int j = 0; j < 4; ++j) {
                int i = tid + j * 128;                       // 512 x 16B transfers
                int dv = i >> 3, part = i & 7;
                const unsigned short* gV =
                    Vt + ((size_t)b * 64 + dv) * T_ + kt + part * 8;
                __builtin_amdgcn_global_load_async_to_lds_b128(
                    GAS(gK + (size_t)i * 8),
                    LAS((char*)Kt_s + i * 16), 0, 0);
                __builtin_amdgcn_global_load_async_to_lds_b128(
                    GAS(gV),
                    LAS((char*)Vt_s + i * 16), 0, 0);
            }
            __builtin_amdgcn_s_wait_asynccnt(0);
#else
            const uint4* src = (const uint4*)gK;
            uint4* dstK = (uint4*)Kt_s;
            uint4* dstV = (uint4*)Vt_s;
            #pragma unroll
            for (int j = 0; j < 4; ++j) {
                int i = tid + j * 128;
                dstK[i] = src[i];
                int dv = i >> 3, part = i & 7;
                dstV[i] = ((const uint4*)(Vt + ((size_t)b * 64 + dv) * T_ + kt))[part];
            }
#endif
        }
        if (kt + 64 < T_) {
            __builtin_prefetch(Kb + ((size_t)b * T_ + kt + 64) * 64, 0, 1);
            __builtin_prefetch(Vt + ((size_t)b * 64 + (tid >> 1)) * T_ + kt + 64, 0, 1);
        }
        __syncthreads();

        #pragma unroll
        for (int sub = 0; sub < 2; ++sub) {
            const int keyb = sub * 32;
            // S = Q * K^T for two 16-key groups (k-dim = 64 -> 2 WMMA each)
            v8f s0 = {}, s1 = {};
            #pragma unroll
            for (int c = 0; c < 2; ++c) {
                bf16x16 b0 = *(const bf16x16*)(Kt_s + (keyb + lp) * 64 + c * 32 + 16 * half);
                s0 = __builtin_amdgcn_wmma_f32_16x16x32_bf16(
                    false, qa[c].v, false, b0, (short)0, s0, false, false);
                bf16x16 b1 = *(const bf16x16*)(Kt_s + (keyb + 16 + lp) * 64 + c * 32 + 16 * half);
                s1 = __builtin_amdgcn_wmma_f32_16x16x32_bf16(
                    false, qa[c].v, false, b1, (short)0, s1, false, false);
            }
            // online softmax over these 32 keys (mask does NOT affect m or l)
            #pragma unroll
            for (int r = 0; r < 8; ++r) {
                float v = rowmax16(fmaxf(s0[r], s1[r]));
                float mnew  = fmaxf(m_r[r], v);
                float alpha = __expf(m_r[r] - mnew);
                m_r[r] = mnew;
                float p0 = __expf(s0[r] - mnew);
                float p1 = __expf(s1[r] - mnew);
                l_r[r] = l_r[r] * alpha + p0 + p1;
                #pragma unroll
                for (int nt = 0; nt < 4; ++nt) o[nt][r] *= alpha;
                // masked-fill AFTER softmax: masked probs become bf16 -inf
                const int row = r + 8 * half;
                const int q   = qbase + row;
                const int sA  = kt + keyb + lp;
                Pw[row * 32 + lp]      = (sA > q)      ? (unsigned short)0xFF80 : f2bf(p0);
                Pw[row * 32 + 16 + lp] = (sA + 16 > q) ? (unsigned short)0xFF80 : f2bf(p1);
            }
            __syncthreads();
            // P (C-layout) -> A-layout via LDS, then O += P * V
            U16 pa;
            pa.qq[0] = *(const uint4*)(Pw + lp * 32 + 16 * half);
            pa.qq[1] = *(const uint4*)(Pw + lp * 32 + 16 * half + 8);
            #pragma unroll
            for (int nt = 0; nt < 4; ++nt) {
                bf16x16 vb = *(const bf16x16*)(Vt_s + (nt * 16 + lp) * 64 + keyb + 16 * half);
                o[nt] = __builtin_amdgcn_wmma_f32_16x16x32_bf16(
                    false, pa.v, false, vb, (short)0, o[nt], false, false);
            }
            __syncthreads();
        }
    }

    // finalize: divide by full-row softmax denominator, scale by 1/sqrt(64)
    #pragma unroll
    for (int r = 0; r < 8; ++r) {
        float l = rowsum16(l_r[r]);
        const float inv = 0.125f / l;
        const int row = qbase + r + 8 * half;
        #pragma unroll
        for (int nt = 0; nt < 4; ++nt)
            out[((size_t)b * T_ + row) * 64 + nt * 16 + lp] = o[nt][r] * inv;
    }
}

// ---------------------------------------------------------------------------
extern "C" void kernel_launch(void* const* d_in, const int* in_sizes, int n_in,
                              void* d_out, int out_size, void* d_ws, size_t ws_size,
                              hipStream_t stream) {
    const float* x  = (const float*)d_in[0];
    const float* Wq = (const float*)d_in[1];
    const float* Wk = (const float*)d_in[2];
    const float* Wv = (const float*)d_in[3];
    // d_in[4] (mask) is a tril buffer; causality is computed from indices.
    float* out = (float*)d_out;

    unsigned short* Wt = (unsigned short*)d_ws;                 // 3 x [64][1024]
    unsigned short* Qb = Wt + 3 * 64 * 1024;                    // [B*T][64]
    unsigned short* Kb = Qb + (size_t)B_ * T_ * 64;             // [B*T][64]
    unsigned short* Vt = Kb + (size_t)B_ * T_ * 64;             // [B][64][T]

    const int wgrid = (64 * 1024 + 255) / 256;
    prep_wt<<<wgrid, 256, 0, stream>>>(Wq, Wt);
    prep_wt<<<wgrid, 256, 0, stream>>>(Wk, Wt + 64 * 1024);
    prep_wt<<<wgrid, 256, 0, stream>>>(Wv, Wt + 2 * 64 * 1024);

    qkv_gemm<0><<<(B_ * T_) / 16, 32, 0, stream>>>(x, Wt,                 Qb);
    qkv_gemm<1><<<(B_ * T_) / 16, 32, 0, stream>>>(x, Wt + 64 * 1024,     Kb);
    qkv_gemm<2><<<(B_ * T_) / 16, 32, 0, stream>>>(x, Wt + 2 * 64 * 1024, Vt);

    attn_kernel<<<dim3(T_ / 64, B_), 128, 0, stream>>>(Qb, Kb, Vt, out);
}
